// DIEN_68659347194011
// MI455X (gfx1250) — matrix operand order repaired and checked by reading, not compiled, not run on hardware
//
#include <hip/hip_runtime.h>
#include <hip/hip_bf16.h>
#include <math.h>

// Problem constants (from reference): B=2048, S=200, D=128
#define B_DIM 2048
#define S_DIM 200
#define D_DIM 128
#define ROWS 16            // batch rows per workgroup (one M-tile)
#define NTHREADS 512       // 16 wave32 waves

typedef __bf16 bf16_t;
typedef __bf16 v4bf  __attribute__((ext_vector_type(4)));
typedef __bf16 v8bf  __attribute__((ext_vector_type(8)));
typedef __bf16 v16bf __attribute__((ext_vector_type(16)));
typedef float  v8f   __attribute__((ext_vector_type(8)));
typedef float  v4f   __attribute__((ext_vector_type(4)));

// ---- LDS layout (bytes) ----
// lw  : 6 weight matrices, bf16, stored transposed [n][k] (read only during init)
// lh  : fp32 h state [16][128]
// lhb : bf16 copy of h for WMMA A-fragments
// lxb : bf16 x_t tile, double buffered
// lz  : fp32 partial pre-activations [16][768]
//       cols: 0-127 x@Wu, 128-255 x@Wr, 256-383 x@Wh, 384-511 h@Uu, 512-639 h@Ur, 640-767 h@Uh
#define LW_ELEMS   (D_DIM * D_DIM)                   // per matrix
#define LW_BYTES   (6 * LW_ELEMS * 2)                // 196608
#define LH_OFF     (LW_BYTES)                        // 196608
#define LHB_OFF    (LH_OFF  + ROWS * D_DIM * 4)      // 204800
#define LXB_OFF    (LHB_OFF + ROWS * D_DIM * 2)      // 208896
#define LZ_OFF     (LXB_OFF + 2 * ROWS * D_DIM * 2)  // 217088
#define SMEM_BYTES (LZ_OFF + ROWS * 768 * 4)         // 266240 (<= 320KB WGP LDS)

__device__ __forceinline__ v8f wmma_bf16(v16bf a, v16bf b, v8f c) {
  // (neg_a, A, neg_b, B, c_mod, C, reuse_a, reuse_b)
  return __builtin_amdgcn_wmma_f32_16x16x32_bf16(false, a, false, b, (short)0, c,
                                                 false, false);
}

// A fragment (16x32 bf16, documented interleaved layout):
// elems 0..7 = K = kb+0..7 ; elems 8..15 = K = kb+16..23, kb = chunk*32 + (lane>=16 ? 8 : 0)
__device__ __forceinline__ v16bf load_a_frag(const bf16_t* p, int row, int kb) {
  union { v16bf v; v8bf h[2]; } u;
  const bf16_t* q = p + row * D_DIM + kb;
  u.h[0] = *(const v8bf*)(q);
  u.h[1] = *(const v8bf*)(q + 16);
  return u.v;
}

// B fragment (32x16 bf16): same interleaved K layout as A, weights stored [n][k] in LDS
__device__ __forceinline__ v16bf load_b_frag(const bf16_t* wt, int ncol, int kb) {
  union { v16bf v; v8bf h[2]; } u;
  const bf16_t* q = wt + ncol * D_DIM + kb;
  u.h[0] = *(const v8bf*)(q);
  u.h[1] = *(const v8bf*)(q + 16);
  return u.v;
}

// Fast gates: v_exp_f32 + v_rcp_f32, no IEEE div fixup chain (1-ulp rcp is far
// below the bf16 GEMM operand noise floor).
__device__ __forceinline__ float sigmoidf_(float z) {
  return __builtin_amdgcn_rcpf(1.0f + __expf(-z));
}

__device__ __forceinline__ float tanhf_(float z) {
#if __has_builtin(__builtin_amdgcn_tanhf)
  return __builtin_amdgcn_tanhf(z);              // gfx1250 V_TANH_F32
#elif __has_builtin(__builtin_amdgcn_tanh_f32)
  return __builtin_amdgcn_tanh_f32(z);           // alt builtin name
#else
  // tanh(z) = 2*sigmoid(2z) - 1; saturates correctly via exp over/underflow
  return 2.0f * __builtin_amdgcn_rcpf(1.0f + __expf(-2.0f * z)) - 1.0f;
#endif
}

__global__ void __launch_bounds__(NTHREADS, 1)
dien_gru_kernel(const float* __restrict__ x,    // [B,S,D]
                const float* __restrict__ h0,   // [B,D]
                const float* __restrict__ Wu, const float* __restrict__ Uu,
                const float* __restrict__ bu,
                const float* __restrict__ Wr, const float* __restrict__ Ur,
                const float* __restrict__ br,
                const float* __restrict__ Wh, const float* __restrict__ Uh,
                const float* __restrict__ bh,
                float* __restrict__ out)        // [B,S,D] ++ [B,D]
{
  extern __shared__ char smem[];
  bf16_t* lw  = (bf16_t*)(smem);
  float*  lh  = (float*) (smem + LH_OFF);
  bf16_t* lhb = (bf16_t*)(smem + LHB_OFF);
  bf16_t* lxb = (bf16_t*)(smem + LXB_OFF);
  float*  lz  = (float*) (smem + LZ_OFF);

  const int tid = threadIdx.x;
  const int b0  = blockIdx.x * ROWS;

  // -------- one-time init: weights -> LDS (bf16, transposed [n][k]) --------
  {
    const float* wsrc[6] = {Wu, Uu, Wr, Ur, Wh, Uh};
#pragma unroll
    for (int m = 0; m < 6; ++m) {
      const float* src = wsrc[m];
      bf16_t* dst = lw + m * LW_ELEMS;
      for (int i = tid; i < LW_ELEMS; i += NTHREADS) {
        int k = i >> 7, n = i & 127;        // src is [k][n] row-major
        dst[n * D_DIM + k] = (bf16_t)src[i];
      }
    }
  }

  // per-thread elementwise mapping: 4 contiguous features of one row
  const int erow = tid >> 5;            // 0..15
  const int d0   = (tid & 31) * 4;      // 0,4,...,124
  const size_t gRow = (size_t)(b0 + erow);

  // biases -> registers
  float bu_r[4], br_r[4], bh_r[4];
#pragma unroll
  for (int j = 0; j < 4; ++j) {
    bu_r[j] = bu[d0 + j];
    br_r[j] = br[d0 + j];
    bh_r[j] = bh[d0 + j];
  }

  // h0 -> LDS (fp32 + bf16 copy)
  {
    v4f a = *(const v4f*)(h0 + gRow * D_DIM + d0);
    v4bf ab;
#pragma unroll
    for (int j = 0; j < 4; ++j) ab[j] = (bf16_t)a[j];
    *(v4f*) (lh  + erow * D_DIM + d0) = a;
    *(v4bf*)(lhb + erow * D_DIM + d0) = ab;
  }

  // x_0 -> LDS buffer 0 (bf16)
  {
    v4f a = *(const v4f*)(x + (gRow * S_DIM + 0) * D_DIM + d0);
    v4bf ab;
#pragma unroll
    for (int j = 0; j < 4; ++j) ab[j] = (bf16_t)a[j];
    *(v4bf*)(lxb + erow * D_DIM + d0) = ab;
  }

  __syncthreads();

  // -------- wave / WMMA mapping (side-specialized waves) --------
  const int lane   = tid & 31;
  const int wv     = tid >> 5;                 // 0..15
  const bool xside = (wv < 8);                 // waves 0-7: x@{Wu,Wr,Wh}; 8-15: h@{Uu,Ur,Uh}
  const int ntile  = xside ? wv : (wv - 8);    // N-tile 0..7
  const int arow   = lane & 15;                // A-fragment M row
  const int khalfA = (lane & 16) ? 8 : 0;      // A interleave half
  const int ncolB  = (ntile << 4) + (lane & 15);
  const int khalfB = (lane & 16) ? 8 : 0;      // B interleave half
  const int mhalf  = (lane & 16) ? 8 : 0;      // C/D row offset
  const int zc0    = xside ? 0   : 384;        // partial-sum columns in lz
  const int zc1    = xside ? 128 : 512;
  const int zc2    = xside ? 256 : 640;

  // Hoist this wave's loop-invariant B-fragments into registers:
  //   xside: Wu, Wr, Wh   |   hside: Uu, Ur, Uh   (3 x 4 chunks x 8 VGPRs = 96 VGPRs)
  const bf16_t* wm0 = lw + (xside ? 0 : 1) * LW_ELEMS;
  const bf16_t* wm1 = lw + (xside ? 2 : 3) * LW_ELEMS;
  const bf16_t* wm2 = lw + (xside ? 4 : 5) * LW_ELEMS;
  v16bf bM0[4], bM1[4], bM2[4];
#pragma unroll
  for (int c = 0; c < 4; ++c) {
    const int kb = c * 32 + khalfB;
    bM0[c] = load_b_frag(wm0, ncolB, kb);
    bM1[c] = load_b_frag(wm1, ncolB, kb);
    bM2[c] = load_b_frag(wm2, ncolB, kb);
  }

  int cur = 0;
  for (int t = 0; t < S_DIM; ++t) {
    const bf16_t* lx_cur = lxb + cur * (ROWS * D_DIM);
    bf16_t*       lx_nxt = lxb + (cur ^ 1) * (ROWS * D_DIM);

    // Each wave reads exactly one A source: x tile (xside) or h tile (hside).
    const bf16_t* pa = xside ? lx_cur : lhb;

    // ---- phase A: preload all A-fragments, then 12 WMMA (3 indep chains) ----
    // Preloading lets LDS latency overlap the WMMA stream (progressive dscnt waits).
    v16bf a[4];
#pragma unroll
    for (int c = 0; c < 4; ++c) a[c] = load_a_frag(pa, arow, c * 32 + khalfA);

    v8f acc0 = {0, 0, 0, 0, 0, 0, 0, 0};
    v8f acc1 = {0, 0, 0, 0, 0, 0, 0, 0};
    v8f acc2 = {0, 0, 0, 0, 0, 0, 0, 0};
#pragma unroll
    for (int c = 0; c < 4; ++c) {
      acc0 = wmma_bf16(a[c], bM0[c], acc0);
      acc1 = wmma_bf16(a[c], bM1[c], acc1);
      acc2 = wmma_bf16(a[c], bM2[c], acc2);
    }
    // scatter C/D fragments (16x16 fp32 layout: VGPR j -> row j+mhalf, lane&15 -> col)
#pragma unroll
    for (int j = 0; j < 8; ++j) {
      float* zrow = lz + (j + mhalf) * 768 + ncolB;
      zrow[zc0] = acc0[j];
      zrow[zc1] = acc1[j];
      zrow[zc2] = acc2[j];
    }

    __syncthreads();

    // ---- phase B: combine partials, gates + state update (fp32), write outs[t] ----
    {
      const float* zr = lz + erow * 768;
      v4f zux = *(const v4f*)(zr + d0);          // x@Wu
      v4f zrx = *(const v4f*)(zr + 128 + d0);    // x@Wr
      v4f zhx = *(const v4f*)(zr + 256 + d0);    // x@Wh
      v4f zuh = *(const v4f*)(zr + 384 + d0);    // h@Uu
      v4f zrh = *(const v4f*)(zr + 512 + d0);    // h@Ur
      v4f zhh = *(const v4f*)(zr + 640 + d0);    // h@Uh
      v4f hp  = *(const v4f*)(lh + erow * D_DIM + d0);
      v4f hn;
      v4bf hnb;
#pragma unroll
      for (int j = 0; j < 4; ++j) {
        float u  = sigmoidf_(zux[j] + zuh[j] + bu_r[j]);
        float r  = sigmoidf_(zrx[j] + zrh[j] + br_r[j]);
        float hh = tanhf_(zhx[j] + r * zhh[j] + bh_r[j]);
        // softmax over singleton axis == 1.0  =>  u_hat == u
        hn[j]  = (1.0f - u) * hp[j] + u * hh;
        hnb[j] = (bf16_t)hn[j];
      }
      *(v4f*) (lh  + erow * D_DIM + d0) = hn;
      *(v4bf*)(lhb + erow * D_DIM + d0) = hnb;
      *(v4f*)(out + (gRow * S_DIM + t) * D_DIM + d0) = hn;
    }

    // stage x_{t+1} into the other LDS buffer; prefetch x_{t+2}
    if (t + 1 < S_DIM) {
      v4f a2 = *(const v4f*)(x + (gRow * S_DIM + (t + 1)) * D_DIM + d0);
      v4bf ab;
#pragma unroll
      for (int j = 0; j < 4; ++j) ab[j] = (bf16_t)a2[j];
      *(v4bf*)(lx_nxt + erow * D_DIM + d0) = ab;
    }
    if (t + 2 < S_DIM) {
      __builtin_prefetch(x + (gRow * S_DIM + (t + 2)) * D_DIM + d0, 0, 0);
    }

    __syncthreads();
    cur ^= 1;
  }

  // ---- h_last (this thread owns exactly the lh elements it last wrote) ----
  *(v4f*)(out + (size_t)B_DIM * S_DIM * D_DIM + gRow * D_DIM + d0) =
      *(const v4f*)(lh + erow * D_DIM + d0);
}

extern "C" void kernel_launch(void* const* d_in, const int* in_sizes, int n_in,
                              void* d_out, int out_size, void* d_ws, size_t ws_size,
                              hipStream_t stream) {
  (void)in_sizes; (void)n_in; (void)out_size; (void)d_ws; (void)ws_size;
  const float* x  = (const float*)d_in[0];
  // d_in[1] = item : unused (softmax over singleton axis == 1.0)
  const float* h0 = (const float*)d_in[2];
  const float* Wu = (const float*)d_in[3];
  const float* Uu = (const float*)d_in[4];
  const float* bu = (const float*)d_in[5];
  const float* Wr = (const float*)d_in[6];
  const float* Ur = (const float*)d_in[7];
  const float* br = (const float*)d_in[8];
  const float* Wh = (const float*)d_in[9];
  const float* Uh = (const float*)d_in[10];
  const float* bh = (const float*)d_in[11];
  // d_in[12] = Wa : unused (dead code in reference)
  float* out = (float*)d_out;

  dien_gru_kernel<<<dim3(B_DIM / ROWS), dim3(NTHREADS), SMEM_BYTES, stream>>>(
      x, h0, Wu, Uu, bu, Wr, Ur, br, Wh, Uh, bh, out);
}
